// _VehicleGNN_45337674776724
// MI455X (gfx1250) — matrix operand order repaired and checked by reading, not compiled, not run on hardware
//
#include <hip/hip_runtime.h>
#include <hip/hip_bf16.h>
#include <stdint.h>

// ---------------- problem constants (match reference) ----------------
#define V_NODES 100000
#define E_EDGES 800000
#define DIM     128      // D
#define KPAD    160      // padded K for both GEMMs (5 k-steps of 32)
#define K_PROJ  138      // D + 1 + K + 1
#define K_EDGE  144      // D + EDGE_FEAT
#define META_W  10       // 1 + K + 1
#define EFEAT   16
#define NVEH    16
#define LN_EPS  1e-5f
#define WPB     4        // waves per block for GEMM kernels (128 threads, wave32)

typedef __bf16        v16bf __attribute__((ext_vector_type(16)));
typedef float         v8f   __attribute__((ext_vector_type(8)));
typedef unsigned int  u32x4 __attribute__((ext_vector_type(4)));

union Frag { v16bf v; u32x4 q[2]; };

__device__ __forceinline__ unsigned short f2bf(float f) {
  // round-to-nearest-even fp32 -> bf16
  unsigned int u = __float_as_uint(f);
  u += 0x7FFFu + ((u >> 16) & 1u);
  return (unsigned short)(u >> 16);
}

// ---------------- elementwise / prep kernels ----------------
__global__ void k_zero_f32(float* p, int n) {
  int i = blockIdx.x * blockDim.x + threadIdx.x;
  if (i < n) p[i] = 0.0f;
}
__global__ void k_zero_u32(unsigned int* p, int n) {
  int i = blockIdx.x * blockDim.x + threadIdx.x;
  if (i < n) p[i] = 0u;
}
__global__ void k_count(const int* __restrict__ edst, unsigned int* cnt, int e) {
  int i = blockIdx.x * blockDim.x + threadIdx.x;
  if (i < e) atomicAdd(&cnt[edst[i]], 1u);
}
// X0[v][k] = bf16( k<128 ? Z[v][k] : k<138 ? meta[v][k-128] : 0 ),  KPAD-padded
__global__ void k_prep_x0(const float* __restrict__ Z, const float* __restrict__ meta,
                          unsigned short* __restrict__ X0) {
  int i = blockIdx.x * blockDim.x + threadIdx.x;
  if (i >= V_NODES * KPAD) return;
  int v = i / KPAD, k = i % KPAD;
  float x = 0.0f;
  if (k < DIM) x = Z[(size_t)v * DIM + k];
  else if (k < K_PROJ) x = meta[(size_t)v * META_W + (k - DIM)];
  X0[i] = f2bf(x);
}
__global__ void k_prep_bf16(const float* __restrict__ src, unsigned short* __restrict__ dst, int n) {
  int i = blockIdx.x * blockDim.x + threadIdx.x;
  if (i < n) dst[i] = f2bf(src[i]);
}
// Wt[n][k] = bf16( k<K ? W[k][n] : 0 )   (N-major so B-fragments are contiguous)
__global__ void k_transpose_bf16(const float* __restrict__ W, unsigned short* __restrict__ Wt,
                                 int K, int N, int Kpad) {
  int i = blockIdx.x * blockDim.x + threadIdx.x;
  if (i >= N * Kpad) return;
  int n = i / Kpad, k = i % Kpad;
  Wt[i] = f2bf(k < K ? W[(size_t)k * N + n] : 0.0f);
}

// ---------------- node projection: h = [Z|meta] @ W0 + b0 ----------------
// one wave per 32-row pair of 16-row M-tiles: every B (weight) fragment feeds
// TWO independent WMMA chains; explicit double buffer on B so waits are not
// full drains. 8 N-tiles x 5 k-steps, fp32 accumulate.
// launch_bounds(128,1): allow full VGPR budget -> no accumulator spills.
__global__ void __launch_bounds__(WPB * 32, 1)
k_node_proj(const unsigned short* __restrict__ X0,
            const unsigned short* __restrict__ W0t,
            const float* __restrict__ bias,
            float* __restrict__ h, unsigned short* __restrict__ hbf,
            int nwaves) {
  int wave = (int)((blockIdx.x * blockDim.x + threadIdx.x) >> 5);
  int lane = threadIdx.x & 31;
  if (wave >= nwaves) return;
  int m = lane & 15, hi = lane >> 4;

  v8f z8 = {};
  v8f acc0[8], acc1[8];
#pragma unroll
  for (int t = 0; t < 8; t++) { acc0[t] = z8; acc1[t] = z8; }

  int r0 = wave * 32 + m;        // row m of tile 0
  int r1 = wave * 32 + 16 + m;   // row m of tile 1
  const unsigned short* ar0 = X0 + (size_t)r0 * KPAD;
  const unsigned short* ar1 = X0 + (size_t)r1 * KPAD;
#pragma unroll
  for (int ks = 0; ks < 5; ks++) {
    Frag a0, a1;
    a0.q[0] = *(const u32x4*)(ar0 + ks * 32 + hi * 8);
    a0.q[1] = *(const u32x4*)(ar0 + ks * 32 + 16 + hi * 8);
    a1.q[0] = *(const u32x4*)(ar1 + ks * 32 + hi * 8);
    a1.q[1] = *(const u32x4*)(ar1 + ks * 32 + 16 + hi * 8);
    const unsigned short* wk = W0t + ks * 32 + hi * 8 + (size_t)m * KPAD;
    Frag bc;                                   // double-buffered B fragment
    bc.q[0] = *(const u32x4*)(wk);
    bc.q[1] = *(const u32x4*)(wk + 16);
#pragma unroll
    for (int nt = 0; nt < 8; nt++) {
      Frag bn;
      if (nt < 7) {
        bn.q[0] = *(const u32x4*)(wk + (size_t)(nt + 1) * 16 * KPAD);
        bn.q[1] = *(const u32x4*)(wk + (size_t)(nt + 1) * 16 * KPAD + 16);
      }
      acc0[nt] = __builtin_amdgcn_wmma_f32_16x16x32_bf16(false, a0.v, false, bc.v,
                                                         (short)0, acc0[nt], false, false);
      acc1[nt] = __builtin_amdgcn_wmma_f32_16x16x32_bf16(false, a1.v, false, bc.v,
                                                         (short)0, acc1[nt], false, false);
      if (nt < 7) bc = bn;
    }
  }
#pragma unroll
  for (int nt = 0; nt < 8; nt++) {
    int n = nt * 16 + m;
    float bv = bias[n];
#pragma unroll
    for (int i = 0; i < 8; i++) {
      // C layout: VGPR i -> row i (+8 for hi lanes)
      int ra = wave * 32 + i + hi * 8;
      int rb_ = wave * 32 + 16 + i + hi * 8;
      float v0 = acc0[nt][i] + bv;
      float v1 = acc1[nt][i] + bv;
      h[(size_t)ra * DIM + n] = v0;
      hbf[(size_t)ra * DIM + n] = f2bf(v0);
      h[(size_t)rb_ * DIM + n] = v1;
      hbf[(size_t)rb_ * DIM + n] = f2bf(v1);
    }
  }
}

// ---------------- edge MLP + scatter-add ----------------
// one wave per 32-edge tile pair: each loaded B (weight) fragment feeds TWO
// independent WMMA chains (tiles t0/t1), halving weight traffic per FLOP and
// doubling in-wave matrix ILP. GEMM1 (K=160) -> relu -> LDS transpose ->
// GEMM2 (K=128) -> atomic scatter into agg[dst].
// launch_bounds(128,1): allow full VGPR budget -> no accumulator spills.
__global__ void __launch_bounds__(WPB * 32, 1)
k_edge(const unsigned short* __restrict__ hbf,
       const unsigned short* __restrict__ eattr,
       const int* __restrict__ esrc, const int* __restrict__ edst,
       const unsigned short* __restrict__ W1t, const float* __restrict__ b1,
       const unsigned short* __restrict__ W2t, const float* __restrict__ b2,
       float* __restrict__ agg, int nwaves) {
  __shared__ unsigned short hid[WPB][2][16 * DIM];   // 2 bf16 16x128 tiles per wave
  int wslot = threadIdx.x >> 5;
  int wave  = blockIdx.x * WPB + wslot;
  int lane  = threadIdx.x & 31;
  if (wave >= nwaves) return;
  int m = lane & 15, hi = lane >> 4;

  int e0 = wave * 32 + m;          // row m of tile 0
  int e1 = wave * 32 + 16 + m;     // row m of tile 1
  int s0 = esrc[e0];
  int s1 = esrc[e1];
  const unsigned short* ar0 = hbf + (size_t)s0 * DIM;
  const unsigned short* ar1 = hbf + (size_t)s1 * DIM;
  __builtin_prefetch(W2t + (size_t)m * DIM, 0, 1);   // warm W2 rows (global_prefetch_b8)

  v8f z8 = {};
  u32x4 zq = {};
  v8f acc0[8], acc1[8];
#pragma unroll
  for (int t = 0; t < 8; t++) { acc0[t] = z8; acc1[t] = z8; }

  // ---- GEMM1: [h_src | e_attr | 0pad] @ W1 ----
#pragma unroll
  for (int ks = 0; ks < 5; ks++) {
    Frag a0, a1;
    if (ks < 4) {
      a0.q[0] = *(const u32x4*)(ar0 + ks * 32 + hi * 8);
      a0.q[1] = *(const u32x4*)(ar0 + ks * 32 + 16 + hi * 8);
      a1.q[0] = *(const u32x4*)(ar1 + ks * 32 + hi * 8);
      a1.q[1] = *(const u32x4*)(ar1 + ks * 32 + 16 + hi * 8);
    } else {  // K=128..143 = edge attrs, K=144..159 = zero pad
      a0.q[0] = *(const u32x4*)(eattr + (size_t)e0 * EFEAT + hi * 8);
      a0.q[1] = zq;
      a1.q[0] = *(const u32x4*)(eattr + (size_t)e1 * EFEAT + hi * 8);
      a1.q[1] = zq;
    }
    const unsigned short* wk = W1t + ks * 32 + hi * 8 + (size_t)m * KPAD;
    Frag bc;                                   // double-buffered B fragment
    bc.q[0] = *(const u32x4*)(wk);
    bc.q[1] = *(const u32x4*)(wk + 16);
#pragma unroll
    for (int nt = 0; nt < 8; nt++) {
      Frag bn;
      if (nt < 7) {
        bn.q[0] = *(const u32x4*)(wk + (size_t)(nt + 1) * 16 * KPAD);
        bn.q[1] = *(const u32x4*)(wk + (size_t)(nt + 1) * 16 * KPAD + 16);
      }
      acc0[nt] = __builtin_amdgcn_wmma_f32_16x16x32_bf16(false, a0.v, false, bc.v,
                                                         (short)0, acc0[nt], false, false);
      acc1[nt] = __builtin_amdgcn_wmma_f32_16x16x32_bf16(false, a1.v, false, bc.v,
                                                         (short)0, acc1[nt], false, false);
      if (nt < 7) bc = bn;
    }
  }

  // ---- bias + relu, store both hidden tiles row-major into LDS ----
#pragma unroll
  for (int nt = 0; nt < 8; nt++) {
    int n = nt * 16 + m;
    float b1v = b1[n];
#pragma unroll
    for (int i = 0; i < 8; i++) {
      float v0 = acc0[nt][i] + b1v; v0 = v0 > 0.0f ? v0 : 0.0f;
      float v1 = acc1[nt][i] + b1v; v1 = v1 > 0.0f ? v1 : 0.0f;
      hid[wslot][0][(i + hi * 8) * DIM + n] = f2bf(v0);
      hid[wslot][1][(i + hi * 8) * DIM + n] = f2bf(v1);
    }
  }
  asm volatile("s_wait_dscnt 0x0" ::: "memory");   // same-wave LDS RAW

  // ---- GEMM2: msg = hidden @ W2 (accumulators reused) ----
#pragma unroll
  for (int t = 0; t < 8; t++) { acc0[t] = z8; acc1[t] = z8; }
#pragma unroll
  for (int ks = 0; ks < 4; ks++) {
    Frag a0, a1;
    a0.q[0] = *(const u32x4*)(&hid[wslot][0][m * DIM + ks * 32 + hi * 8]);
    a0.q[1] = *(const u32x4*)(&hid[wslot][0][m * DIM + ks * 32 + 16 + hi * 8]);
    a1.q[0] = *(const u32x4*)(&hid[wslot][1][m * DIM + ks * 32 + hi * 8]);
    a1.q[1] = *(const u32x4*)(&hid[wslot][1][m * DIM + ks * 32 + 16 + hi * 8]);
    const unsigned short* wk = W2t + ks * 32 + hi * 8 + (size_t)m * DIM;
    Frag bc;
    bc.q[0] = *(const u32x4*)(wk);
    bc.q[1] = *(const u32x4*)(wk + 16);
#pragma unroll
    for (int nt = 0; nt < 8; nt++) {
      Frag bn;
      if (nt < 7) {
        bn.q[0] = *(const u32x4*)(wk + (size_t)(nt + 1) * 16 * DIM);
        bn.q[1] = *(const u32x4*)(wk + (size_t)(nt + 1) * 16 * DIM + 16);
      }
      acc0[nt] = __builtin_amdgcn_wmma_f32_16x16x32_bf16(false, a0.v, false, bc.v,
                                                         (short)0, acc0[nt], false, false);
      acc1[nt] = __builtin_amdgcn_wmma_f32_16x16x32_bf16(false, a1.v, false, bc.v,
                                                         (short)0, acc1[nt], false, false);
      if (nt < 7) bc = bn;
    }
  }

  // ---- scatter-add (agg is L2-resident: 51 MB << 192 MB L2) ----
  int d0[8], d1[8];
#pragma unroll
  for (int i = 0; i < 8; i++) {
    d0[i] = edst[wave * 32 + i + hi * 8];
    d1[i] = edst[wave * 32 + 16 + i + hi * 8];
  }
#pragma unroll
  for (int nt = 0; nt < 8; nt++) {
    int n = nt * 16 + m;
    float b2v = b2[n];
#pragma unroll
    for (int i = 0; i < 8; i++) {
      atomicAdd(&agg[(size_t)d0[i] * DIM + n], acc0[nt][i] + b2v);
      atomicAdd(&agg[(size_t)d1[i] * DIM + n], acc1[nt][i] + b2v);
    }
  }
}

// ---------------- h = LN(h + agg/count); refresh bf16 copy ----------------
__global__ void k_update(float* __restrict__ h, unsigned short* __restrict__ hbf,
                         const float* __restrict__ agg, const unsigned int* __restrict__ cnt,
                         const float* __restrict__ g, const float* __restrict__ b) {
  int v = (int)((blockIdx.x * blockDim.x + threadIdx.x) >> 5);
  int lane = threadIdx.x & 31;
  if (v >= V_NODES) return;
  unsigned int c = cnt[v];
  float inv = 1.0f / (float)(c ? c : 1u);
  float x[4], s = 0.0f;
#pragma unroll
  for (int t = 0; t < 4; t++) {
    int j = lane + 32 * t;
    x[t] = h[(size_t)v * DIM + j] + agg[(size_t)v * DIM + j] * inv;
    s += x[t];
  }
#pragma unroll
  for (int o = 16; o >= 1; o >>= 1) s += __shfl_xor(s, o, 32);
  float mean = s * (1.0f / 128.0f);
  float vs = 0.0f;
#pragma unroll
  for (int t = 0; t < 4; t++) { float d = x[t] - mean; vs += d * d; }
#pragma unroll
  for (int o = 16; o >= 1; o >>= 1) vs += __shfl_xor(vs, o, 32);
  float r = rsqrtf(vs * (1.0f / 128.0f) + LN_EPS);
#pragma unroll
  for (int t = 0; t < 4; t++) {
    int j = lane + 32 * t;
    float y = (x[t] - mean) * r * g[j] + b[j];
    h[(size_t)v * DIM + j] = y;
    hbf[(size_t)v * DIM + j] = f2bf(y);
  }
}

// ---------------- readout: LN(h[cur] + propsMLP(veh)) ----------------
__global__ void k_readout(const float* __restrict__ h, const int* __restrict__ cur,
                          const float* __restrict__ vf,
                          const float* __restrict__ pw1, const float* __restrict__ pb1,
                          const float* __restrict__ pw2, const float* __restrict__ pb2,
                          const float* __restrict__ rg, const float* __restrict__ rb,
                          float* __restrict__ out) {
  __shared__ float hidden[NVEH][DIM];
  int w = threadIdx.x >> 5, lane = threadIdx.x & 31;
  float f[8];
#pragma unroll
  for (int k = 0; k < 8; k++) f[k] = vf[w * 8 + k];
#pragma unroll
  for (int t = 0; t < 4; t++) {
    int jh = lane + 32 * t;
    float s = pb1[jh];
#pragma unroll
    for (int k = 0; k < 8; k++) s += f[k] * pw1[k * DIM + jh];
    hidden[w][jh] = fmaxf(s, 0.0f);
  }
  asm volatile("s_wait_dscnt 0x0" ::: "memory");   // same-wave LDS RAW
  int node = cur[w];
  float x[4], s = 0.0f;
#pragma unroll
  for (int t = 0; t < 4; t++) {
    int j = lane + 32 * t;
    float o = pb2[j];
    for (int k = 0; k < DIM; k++) o += hidden[w][k] * pw2[k * DIM + j];
    x[t] = o + h[(size_t)node * DIM + j];
    s += x[t];
  }
#pragma unroll
  for (int o = 16; o >= 1; o >>= 1) s += __shfl_xor(s, o, 32);
  float mean = s * (1.0f / 128.0f);
  float vs = 0.0f;
#pragma unroll
  for (int t = 0; t < 4; t++) { float d = x[t] - mean; vs += d * d; }
#pragma unroll
  for (int o = 16; o >= 1; o >>= 1) vs += __shfl_xor(vs, o, 32);
  float r = rsqrtf(vs * (1.0f / 128.0f) + LN_EPS);
#pragma unroll
  for (int t = 0; t < 4; t++) {
    int j = lane + 32 * t;
    out[w * DIM + j] = (x[t] - mean) * r * rg[j] + rb[j];
  }
}

// ---------------- host orchestration ----------------
extern "C" void kernel_launch(void* const* d_in, const int* in_sizes, int n_in,
                              void* d_out, int out_size, void* d_ws, size_t ws_size,
                              hipStream_t stream) {
  const float* Z      = (const float*)d_in[0];
  const float* meta   = (const float*)d_in[1];
  const float* vf     = (const float*)d_in[2];
  const float* ea     = (const float*)d_in[3];
  const int*   esrc   = (const int*)d_in[4];
  const int*   edst   = (const int*)d_in[5];
  const int*   cur    = (const int*)d_in[6];
  const float* w0     = (const float*)d_in[7];
  const float* b0     = (const float*)d_in[8];
  const float* lw1    = (const float*)d_in[9];
  const float* lb1    = (const float*)d_in[10];
  const float* lw2    = (const float*)d_in[11];
  const float* lb2    = (const float*)d_in[12];
  const float* lng    = (const float*)d_in[13];
  const float* lnb    = (const float*)d_in[14];
  const float* pw1    = (const float*)d_in[15];
  const float* pb1    = (const float*)d_in[16];
  const float* pw2    = (const float*)d_in[17];
  const float* pb2    = (const float*)d_in[18];
  const float* rg     = (const float*)d_in[19];
  const float* rb     = (const float*)d_in[20];
  float* out          = (float*)d_out;

  char* base = (char*)d_ws;
  size_t off = 0;
#define TAKE(T, NAME, NBYTES) T NAME = (T)(base + off); off += (((size_t)(NBYTES)) + 255) & ~(size_t)255;
  TAKE(unsigned short*, X0,   (size_t)V_NODES * KPAD * 2)
  TAKE(unsigned short*, EATT, (size_t)E_EDGES * EFEAT * 2)
  TAKE(float*,          H,    (size_t)V_NODES * DIM * 4)
  TAKE(unsigned short*, HBF,  (size_t)V_NODES * DIM * 2)
  TAKE(float*,          AGG,  (size_t)V_NODES * DIM * 4)
  TAKE(unsigned int*,   CNT,  (size_t)V_NODES * 4)
  TAKE(unsigned short*, W0T,  (size_t)DIM * KPAD * 2)
  TAKE(unsigned short*, W1T,  (size_t)2 * DIM * KPAD * 2)
  TAKE(unsigned short*, W2T,  (size_t)2 * DIM * DIM * 2)
#undef TAKE

  const int T = 256;
  // counts
  k_zero_u32<<<(V_NODES + T - 1) / T, T, 0, stream>>>(CNT, V_NODES);
  k_count<<<(E_EDGES + T - 1) / T, T, 0, stream>>>(edst, CNT, E_EDGES);
  // bf16 prep
  k_prep_x0<<<(V_NODES * KPAD + T - 1) / T, T, 0, stream>>>(Z, meta, X0);
  k_prep_bf16<<<(E_EDGES * EFEAT + T - 1) / T, T, 0, stream>>>(ea, EATT, E_EDGES * EFEAT);
  k_transpose_bf16<<<(DIM * KPAD + T - 1) / T, T, 0, stream>>>(w0, W0T, K_PROJ, DIM, KPAD);
  for (int l = 0; l < 2; l++) {
    k_transpose_bf16<<<(DIM * KPAD + T - 1) / T, T, 0, stream>>>(
        lw1 + (size_t)l * K_EDGE * DIM, W1T + (size_t)l * DIM * KPAD, K_EDGE, DIM, KPAD);
    k_transpose_bf16<<<(DIM * DIM + T - 1) / T, T, 0, stream>>>(
        lw2 + (size_t)l * DIM * DIM, W2T + (size_t)l * DIM * DIM, DIM, DIM, DIM);
  }
  // node projection (two 16-row tiles per wave)
  int nwave_v = V_NODES / 32;  // 3125
  k_node_proj<<<(nwave_v + WPB - 1) / WPB, WPB * 32, 0, stream>>>(X0, W0T, b0, H, HBF, nwave_v);
  // GNN layers
  int nwave_e = E_EDGES / 32;  // 25000 (two 16-edge tiles per wave)
  for (int l = 0; l < 2; l++) {
    k_zero_f32<<<(V_NODES * DIM + T - 1) / T, T, 0, stream>>>(AGG, V_NODES * DIM);
    k_edge<<<(nwave_e + WPB - 1) / WPB, WPB * 32, 0, stream>>>(
        HBF, EATT, esrc, edst,
        W1T + (size_t)l * DIM * KPAD, lb1 + (size_t)l * DIM,
        W2T + (size_t)l * DIM * DIM,  lb2 + (size_t)l * DIM,
        AGG, nwave_e);
    k_update<<<((V_NODES * 32) + T - 1) / T, T, 0, stream>>>(
        H, HBF, AGG, CNT, lng + (size_t)l * DIM, lnb + (size_t)l * DIM);
  }
  // readout
  k_readout<<<1, NVEH * 32, 0, stream>>>(H, cur, vf, pw1, pb1, pw2, pb2, rg, rb, out);
  (void)in_sizes; (void)n_in; (void)out_size; (void)ws_size;
}
